// SupConLoss_7284264534304
// MI455X (gfx1250) — matrix operand order repaired and checked by reading, not compiled, not run on hardware
//
#include <hip/hip_runtime.h>
#include <hip/hip_bf16.h>

typedef __attribute__((ext_vector_type(16))) _Float16 v16h;
typedef __attribute__((ext_vector_type(8)))  _Float16 v8h;
typedef __attribute__((ext_vector_type(4)))  _Float16 v4h;
typedef __attribute__((ext_vector_type(8)))  float    v8f;

#define BATCH 4096
#define NVIEW 2
#define DDIM  128
#define NTOT  (BATCH * NVIEW)   // 8192
#define NT16  (NTOT / 16)       // 512 column tiles
#define INV_TEMP (1.0f / 0.07f)

// ---------------------------------------------------------------------------
// Kernel 1: L2-normalize each (b,v) feature row, convert to f16, write cf in
// view-major order: cf[v*BATCH + b, :] = normalize(features[b, v, :]).
// One wave (32 lanes) per row; each lane handles 4 contiguous floats.
// ---------------------------------------------------------------------------
__global__ void supcon_normalize_kernel(const float* __restrict__ feat,
                                        _Float16* __restrict__ cf) {
    int gwave = (blockIdx.x * blockDim.x + threadIdx.x) >> 5;  // row n in cf
    int lane  = threadIdx.x & 31;
    if (gwave >= NTOT) return;
    int v = gwave / BATCH;
    int b = gwave - v * BATCH;
    const float* src = feat + ((size_t)b * NVIEW + v) * DDIM;

    float4 x = ((const float4*)src)[lane];
    float ss = x.x * x.x + x.y * x.y + x.z * x.z + x.w * x.w;
#pragma unroll
    for (int off = 16; off > 0; off >>= 1)
        ss += __shfl_xor(ss, off, 32);
    float inv = 1.0f / fmaxf(sqrtf(ss), 1e-8f);

    v4h h;
    h.x = (_Float16)(x.x * inv);
    h.y = (_Float16)(x.y * inv);
    h.z = (_Float16)(x.z * inv);
    h.w = (_Float16)(x.w * inv);
    *(v4h*)(cf + (size_t)gwave * DDIM + lane * 4) = h;
}

// ---------------------------------------------------------------------------
// Kernel 2: zero the scalar output (harness poisons d_out with 0xAA).
// ---------------------------------------------------------------------------
__global__ void supcon_zero_kernel(float* __restrict__ out) {
    if (threadIdx.x == 0 && blockIdx.x == 0) out[0] = 0.0f;
}

// ---------------------------------------------------------------------------
// Kernel 3: fused sim = cf @ cf^T / TEMP (clipped) + masked row reductions.
// blockDim = 256 (8 waves). Wave w owns the 16-row tile
// rowBase = (blockIdx.x*8 + w)*16 and sweeps all 512 column tiles.
// A fragments (K=128) preloaded once; B fragments register-double-buffered.
// Epilogue is fully branchless; diagonal exclusion hoisted to the single
// uniform tile where colBase == rowBase.
// ---------------------------------------------------------------------------
__global__ void __launch_bounds__(256)
supcon_main_kernel(const _Float16* __restrict__ cf,
                   const long long* __restrict__ labels,
                   float* __restrict__ out) {
    __shared__ int s_lab[BATCH];   // 16 KB of the 320 KB WGP LDS

    const int lane    = threadIdx.x & 31;
    const int wave    = threadIdx.x >> 5;
    const int lm      = lane & 15;       // col-within-tile (B/C/D), row-within-tile (A)
    const int hsel    = lane >> 4;       // lane half select
    const int rowBase = (blockIdx.x * 8 + wave) * 16;

    // Stage labels (int32) into LDS once per block.
    for (int i = threadIdx.x; i < BATCH; i += 256)
        s_lab[i] = (int)labels[i];
    __syncthreads();

    // ---- preload A fragments for the full K=128 (invariant over col loop) --
    // A 16x32 in contiguous k order (not the ISA table order): lanes 0-15 hold K=0..15,
    // lanes 16-31 hold K=16..31 of their row (matches B).
    const _Float16* aRow = cf + (size_t)(rowBase + lm) * DDIM;
    v16h afrag[4];
#pragma unroll
    for (int ks = 0; ks < 4; ++ks) {
        int k0 = ks * 32;
        v8h lo = *(const v8h*)(aRow + k0 + 16 * hsel);
        v8h hi = *(const v8h*)(aRow + k0 + 8 + 16 * hsel);
        afrag[ks] = __builtin_shufflevector(lo, hi,
            0, 1, 2, 3, 4, 5, 6, 7, 8, 9, 10, 11, 12, 13, 14, 15);
    }

    // Loop-invariant per-lane state.
    int   labRow[8];
    float keep[8];   // 0 on the diagonal element of the diag tile, else 1
#pragma unroll
    for (int r = 0; r < 8; ++r) {
        labRow[r] = s_lab[(rowBase + r + 8 * hsel) & (BATCH - 1)];
        keep[r]   = (lm != r + 8 * hsel) ? 1.0f : 0.0f;
    }

    float rowsum[8], possum[8], poscnt[8];
#pragma unroll
    for (int r = 0; r < 8; ++r) { rowsum[r] = 0.0f; possum[r] = 0.0f; poscnt[r] = 0.0f; }

    // ---- B fragment register double buffer -------------------------------
    // B 32x16 in contiguous k order (same as A): lane's B column = cf row (colBase+lm);
    // lanes 0-15 hold K=0..15, lanes 16-31 hold K=16..31 (contiguous 32B).
    v16h bbuf[2][4];
    {
        const _Float16* bRow0 = cf + (size_t)lm * DDIM;
#pragma unroll
        for (int ks = 0; ks < 4; ++ks)
            bbuf[0][ks] = *(const v16h*)(bRow0 + ks * 32 + 16 * hsel);
    }

#pragma unroll 2
    for (int jt = 0; jt < NT16; ++jt) {
        const int cur = jt & 1;
        const int nxt = cur ^ 1;
        const int colBase = jt * 16;
        const int nCol    = colBase + lm;

        // Issue next tile's loads before consuming the current fragments.
        const int jn = (jt + 1) & (NT16 - 1);
        const _Float16* bRowN = cf + (size_t)(jn * 16 + lm) * DDIM;
#pragma unroll
        for (int ks = 0; ks < 4; ++ks)
            bbuf[nxt][ks] = *(const v16h*)(bRowN + ks * 32 + 16 * hsel);

        // Warm L0 two tiles ahead.
        const int jp = (jt + 2) & (NT16 - 1);
        __builtin_prefetch(cf + (size_t)(jp * 16 + lm) * DDIM, 0, 3);

        const int labCol = s_lab[nCol & (BATCH - 1)];

        v8f acc = {};
#pragma unroll
        for (int ks = 0; ks < 4; ++ks)
            acc = __builtin_amdgcn_wmma_f32_16x16x32_f16(
                false, afrag[ks], false, bbuf[cur][ks], (short)0, acc, false, false);

        if (colBase != rowBase) {
            // Common path: no diagonal element possible -> fully branchless.
#pragma unroll
            for (int r = 0; r < 8; ++r) {
                float s = fminf(fmaxf(acc[r] * INV_TEMP, -20.0f), 20.0f);
                float m = (labRow[r] == labCol) ? 1.0f : 0.0f;
                rowsum[r] += __expf(s);
                possum[r]  = fmaf(m, s, possum[r]);
                poscnt[r] += m;
            }
        } else {
            // Diagonal tile (exactly one per wave): apply precomputed keep[].
#pragma unroll
            for (int r = 0; r < 8; ++r) {
                float s = fminf(fmaxf(acc[r] * INV_TEMP, -20.0f), 20.0f);
                float m = ((labRow[r] == labCol) ? 1.0f : 0.0f) * keep[r];
                rowsum[r] = fmaf(keep[r], __expf(s), rowsum[r]);
                possum[r] = fmaf(m, s, possum[r]);
                poscnt[r] += m;
            }
        }
    }

    // Reduce across the 16 lanes holding the 16 columns of each row.
    // xor offsets 1,2,4,8 keep lanes 0-15 and 16-31 independent (wave32).
#pragma unroll
    for (int r = 0; r < 8; ++r) {
#pragma unroll
        for (int off = 1; off < 16; off <<= 1) {
            rowsum[r] += __shfl_xor(rowsum[r], off, 32);
            possum[r] += __shfl_xor(possum[r], off, 32);
            poscnt[r] += __shfl_xor(poscnt[r], off, 32);
        }
    }

    if (lm == 0) {  // lanes 0 and 16: rows rowBase+0..7 and rowBase+8..15
        float local = 0.0f;
#pragma unroll
        for (int r = 0; r < 8; ++r) {
            float mlpp = (possum[r] - poscnt[r] * __logf(rowsum[r] + 1e-8f))
                         / (poscnt[r] + 1e-8f);
            local += -mlpp;   // TEMP/BASE_TEMP == 1
        }
        atomicAdd(out, local * (1.0f / (float)NTOT));
    }
}

// ---------------------------------------------------------------------------
extern "C" void kernel_launch(void* const* d_in, const int* in_sizes, int n_in,
                              void* d_out, int out_size, void* d_ws, size_t ws_size,
                              hipStream_t stream) {
    const float*     feat   = (const float*)d_in[0];      // [4096, 2, 128] fp32
    const long long* labels = (const long long*)d_in[1];  // [4096] int64
    float*           out    = (float*)d_out;              // scalar fp32
    _Float16*        cf     = (_Float16*)d_ws;            // [8192, 128] f16 (2 MB)

    supcon_normalize_kernel<<<NTOT / 8, 256, 0, stream>>>(feat, cf);
    supcon_zero_kernel<<<1, 32, 0, stream>>>(out);
    supcon_main_kernel<<<NTOT / 128, 256, 0, stream>>>(cf, labels, out);
}
